// Decoder_11914239279429
// MI455X (gfx1250) — matrix operand order repaired and checked by reading, not compiled, not run on hardware
//
#include <hip/hip_runtime.h>
#include <hip/hip_bf16.h>

// Problem constants (from reference)
#define BB   32
#define TT   22
#define PP   196     // 14*14
#define DEE  2048
#define EEE  512
#define HHH  512
#define AAA  512
#define VVV  30000
#define TDEC 20      // T - 2
#define XXX  (EEE + DEE)   // 2560
#define G4   (4 * HHH)     // 2048

typedef __attribute__((ext_vector_type(16))) __bf16 bf16x16;
typedef __attribute__((ext_vector_type(8)))  float  f32x8;

__device__ __forceinline__ float sigf(float x) { return 1.f / (1.f + __expf(-x)); }

// ---------------------------------------------------------------------------
// Stable descending argsort of caption_lengths (B=32) + sorted caps/lengths,
// also writes the int-valued outputs (as floats) to the output tail.
// ---------------------------------------------------------------------------
__global__ void k_sort(const int* __restrict__ caplen, const int* __restrict__ captions,
                       int* __restrict__ sind, int* __restrict__ slen, int* __restrict__ caps_s,
                       float* __restrict__ out_caps, float* __restrict__ out_len,
                       float* __restrict__ out_sind) {
    if (blockIdx.x != 0 || threadIdx.x != 0) return;
    for (int i = 0; i < BB; ++i) {
        int li = caplen[i], r = 0;
        for (int j = 0; j < BB; ++j) {
            int lj = caplen[j];
            if (lj > li || (lj == li && j < i)) ++r;
        }
        sind[r] = i;
    }
    for (int r = 0; r < BB; ++r) {
        int o = sind[r];
        slen[r]     = caplen[o];
        out_sind[r] = (float)o;
        out_len[r]  = (float)caplen[o];
        for (int t = 0; t < TT; ++t) {
            int cv = captions[t * BB + o];   // captions is [T,B]
            caps_s[r * TT + t]   = cv;
            out_caps[r * TT + t] = (float)cv;
        }
    }
}

// enc_sorted[b] = features[sort_ind[b]]
__global__ void k_gather_enc(const float* __restrict__ feat, const int* __restrict__ sind,
                             float* __restrict__ enc) {
    size_t idx = (size_t)blockIdx.x * blockDim.x + threadIdx.x;
    const size_t per = (size_t)PP * DEE;
    if (idx >= (size_t)BB * per) return;
    int b = (int)(idx / per);
    size_t rest = idx % per;
    enc[idx] = feat[(size_t)sind[b] * per + rest];
}

// mean over pixels
__global__ void k_mean(const float* __restrict__ enc, float* __restrict__ meanE) {
    int b = blockIdx.x;
    for (int d = threadIdx.x; d < DEE; d += blockDim.x) {
        const float* p = enc + (size_t)b * PP * DEE + d;
        float s = 0.f;
        for (int q = 0; q < PP; ++q) s += p[(size_t)q * DEE];
        meanE[b * DEE + d] = s * (1.f / (float)PP);
    }
}

// fp32 -> bf16 straight convert
__global__ void k_cvt(const float* __restrict__ src, __hip_bfloat16* __restrict__ dst, int n) {
    int i = blockIdx.x * blockDim.x + threadIdx.x;
    if (i < n) dst[i] = __float2bfloat16(src[i]);
}

// fp32 [R,C] -> bf16 [C,R] (weights stored [N,K] for contiguous B-fragment loads)
__global__ void k_tcvt(const float* __restrict__ src, __hip_bfloat16* __restrict__ dst,
                       int R, int C) {
    int i = blockIdx.x * blockDim.x + threadIdx.x;
    if (i >= R * C) return;
    int r = i / C, c = i % C;
    dst[(size_t)c * R + r] = __float2bfloat16(src[i]);
}

// ---------------------------------------------------------------------------
// Pack fp32 activations [M,K] into bf16 WMMA A-fragment layout:
//   out[ ((mt*(K/32) + ks)*32 + lane)*16 + j ]
//     = bf16( A[ (mt*16 + lane%16)*lda + ks*32 + klocal(j,lane) ] )
// with klocal contiguous (lanes 0-15: K {0..15},
// lanes 16-31: K {16..31}).  Each lane's 16 bf16 = 32 contiguous
// bytes -> GEMM A-fragment loads are two coalesced b128s.
// ---------------------------------------------------------------------------
__global__ void k_packA(const float* __restrict__ A, int lda,
                        __hip_bfloat16* __restrict__ out, int M, int K) {
    int idx = blockIdx.x * blockDim.x + threadIdx.x;
    if (idx >= M * K) return;
    int j    = idx & 15;
    int lane = (idx >> 4) & 31;
    int rest = idx >> 9;
    int KS   = K >> 5;
    int ks   = rest % KS;
    int mt   = rest / KS;
    int half = lane >> 4;
    int row  = (mt << 4) + (lane & 15);
    int kl   = (half << 4) + j;
    out[idx] = __float2bfloat16(A[(size_t)row * lda + (ks << 5) + kl]);
}

// epilogue: write one 16x16 C tile from an accumulator
__device__ __forceinline__ void store_tile(f32x8 acc, float* __restrict__ C, long long ldc,
                                           int mt, int half, int col,
                                           const float* __restrict__ bias, int act, int accum,
                                           const int* __restrict__ slen, int t) {
    float bv = bias ? bias[col] : 0.f;
#pragma unroll
    for (int r = 0; r < 8; ++r) {
        int row = (mt << 4) + (half << 3) + r;
        float v = acc[r] + bv;
        if (accum) v += C[(size_t)row * ldc + col];
        if (act == 1)      v = fmaxf(v, 0.f);
        else if (act == 2) v = sigf(v);
        if (slen && (slen[row] - 1) <= t) v = 0.f;   // shrinking-batch mask
        C[(size_t)row * ldc + col] = v;
    }
}

// ---------------------------------------------------------------------------
// bf16 WMMA GEMM with pre-packed A fragments and [N,K] bf16 weights.
// One wave computes a 16(M) x 64(N) strip: 4 named accumulators sharing one
// A fragment per K-step.  The K-loop body is STRAIGHT-LINE (no exec-mask
// branching -> no accumulator spills): tail tiles use clamped B pointers and
// compute harmlessly; validity is only applied at the epilogue stores.
// ---------------------------------------------------------------------------
__global__ void k_gemm(const __hip_bfloat16* __restrict__ Apack,
                       const __hip_bfloat16* __restrict__ Bt,
                       const float* __restrict__ bias,
                       float* __restrict__ C, long long ldc,
                       int M, int N, int K, int act, int accum,
                       const int* __restrict__ slen, int t) {
    int ntg  = (N + 63) >> 6;                 // groups of 4 N-tiles
    int tile = blockIdx.x * 8 + (threadIdx.x >> 5);
    if (tile >= (M >> 4) * ntg) return;
    int mt   = tile / ntg;
    int tn0  = (tile % ntg) << 6;
    int lane = threadIdx.x & 31;
    int half = lane >> 4;
    int l15  = lane & 15;
    int KS   = K >> 5;

    bool v1 = (tn0 + 16) < N;
    bool v2 = (tn0 + 32) < N;
    bool v3 = (tn0 + 48) < N;

    const __hip_bfloat16* Ap = Apack + ((size_t)mt * KS * 32 + lane) * 16;
    const __hip_bfloat16* B0 = Bt + (size_t)(tn0 + l15) * K + (half << 4);
    const __hip_bfloat16* B1 = Bt + (size_t)((v1 ? tn0 + 16 : 0) + l15) * K + (half << 4);
    const __hip_bfloat16* B2 = Bt + (size_t)((v2 ? tn0 + 32 : 0) + l15) * K + (half << 4);
    const __hip_bfloat16* B3 = Bt + (size_t)((v3 ? tn0 + 48 : 0) + l15) * K + (half << 4);

    f32x8 acc0 = {}, acc1 = {}, acc2 = {}, acc3 = {};
    for (int ks = 0; ks < KS; ++ks) {
        bf16x16 a = *reinterpret_cast<const bf16x16*>(Ap + (size_t)ks * 512);
        int kb = ks << 5;
        bf16x16 b0 = *reinterpret_cast<const bf16x16*>(B0 + kb);
        bf16x16 b1 = *reinterpret_cast<const bf16x16*>(B1 + kb);
        bf16x16 b2 = *reinterpret_cast<const bf16x16*>(B2 + kb);
        bf16x16 b3 = *reinterpret_cast<const bf16x16*>(B3 + kb);
        acc0 = __builtin_amdgcn_wmma_f32_16x16x32_bf16(false, a, false, b0, (short)0, acc0, false, false);
        acc1 = __builtin_amdgcn_wmma_f32_16x16x32_bf16(false, a, false, b1, (short)0, acc1, false, false);
        acc2 = __builtin_amdgcn_wmma_f32_16x16x32_bf16(false, a, false, b2, (short)0, acc2, false, false);
        acc3 = __builtin_amdgcn_wmma_f32_16x16x32_bf16(false, a, false, b3, (short)0, acc3, false, false);
    }

    store_tile(acc0, C, ldc, mt, half, tn0 + l15, bias, act, accum, slen, t);
    if (v1) store_tile(acc1, C, ldc, mt, half, tn0 + 16 + l15, bias, act, accum, slen, t);
    if (v2) store_tile(acc2, C, ldc, mt, half, tn0 + 32 + l15, bias, act, accum, slen, t);
    if (v3) store_tile(acc3, C, ldc, mt, half, tn0 + 48 + l15, bias, act, accum, slen, t);
}

// e[b,p] = relu(att1[b,p,:] + att2[b,:]) . Wfa + bfa   (one wave per (b,p))
__global__ void k_escore(const float* __restrict__ att1, const float* __restrict__ att2,
                         const float* __restrict__ Wfa, const float* __restrict__ bfa,
                         float* __restrict__ e) {
    int idx = blockIdx.x * 8 + (threadIdx.x >> 5);
    if (idx >= BB * PP) return;
    int b = idx / PP;
    int lane = threadIdx.x & 31;
    const float* a1 = att1 + (size_t)idx * AAA;
    const float* a2 = att2 + (size_t)b * AAA;
    float s = 0.f;
    for (int k = lane; k < AAA; k += 32) {
        float v = fmaxf(a1[k] + a2[k], 0.f);
        s += v * Wfa[k];
    }
    for (int off = 16; off; off >>= 1) s += __shfl_xor(s, off, 32);
    if (lane == 0) e[idx] = s + bfa[0];
}

// per-b softmax over P, masked alpha output, awe = sum_p alpha * enc
__global__ void k_softmax_awe(const float* __restrict__ e, const float* __restrict__ enc,
                              float* __restrict__ awe, float* __restrict__ outA,
                              const int* __restrict__ slen, int t) {
    __shared__ float sh[256];
    __shared__ float sal[PP];
    int b = blockIdx.x, tid = threadIdx.x;

    float v = (tid < PP) ? e[b * PP + tid] : -3.4e38f;
    sh[tid] = v; __syncthreads();
    for (int s = 128; s; s >>= 1) {
        if (tid < s) sh[tid] = fmaxf(sh[tid], sh[tid + s]);
        __syncthreads();
    }
    float mx = sh[0]; __syncthreads();

    float ex = (tid < PP) ? __expf(v - mx) : 0.f;
    sh[tid] = ex; __syncthreads();
    for (int s = 128; s; s >>= 1) {
        if (tid < s) sh[tid] += sh[tid + s];
        __syncthreads();
    }
    float inv = 1.f / sh[0];

    if (tid < PP) {
        float al = ex * inv;
        sal[tid] = al;
        bool active = (slen[b] - 1) > t;
        outA[(size_t)b * TDEC * PP + (size_t)t * PP + tid] = active ? al : 0.f;
    }
    __syncthreads();

    const float* ep = enc + (size_t)b * PP * DEE;
    for (int d = tid; d < DEE; d += blockDim.x) {
        float s = 0.f;
        for (int p = 0; p < PP; ++p) s += sal[p] * ep[(size_t)p * DEE + d];
        awe[b * DEE + d] = s;
    }
}

// x = [ emb[caps[b,t]] , gate * awe ]
__global__ void k_build_x(const int* __restrict__ caps_s, const float* __restrict__ emb,
                          const float* __restrict__ gate, const float* __restrict__ awe,
                          float* __restrict__ x, int t) {
    int idx = blockIdx.x * blockDim.x + threadIdx.x;
    if (idx >= BB * XXX) return;
    int b = idx / XXX, j = idx % XXX;
    float v;
    if (j < EEE) v = emb[(size_t)caps_s[b * TT + t] * EEE + j];
    else {
        int k = j - EEE;
        v = gate[b * DEE + k] * awe[b * DEE + k];
    }
    x[idx] = v;
}

// LSTM cell + masked state update; hnew kept unmasked for the preds GEMM
__global__ void k_lstm(const float* __restrict__ gates, float* __restrict__ h,
                       float* __restrict__ c, float* __restrict__ hnew,
                       const int* __restrict__ slen, int t) {
    int idx = blockIdx.x * blockDim.x + threadIdx.x;
    if (idx >= BB * HHH) return;
    int b = idx / HHH, j = idx % HHH;
    const float* g = gates + (size_t)b * G4;
    float gi = g[j], gf = g[HHH + j], gg = g[2 * HHH + j], go = g[3 * HHH + j];
    float cn = sigf(gf) * c[idx] + sigf(gi) * tanhf(gg);
    float hn = sigf(go) * tanhf(cn);
    hnew[idx] = hn;
    if ((slen[b] - 1) > t) { h[idx] = hn; c[idx] = cn; }
}

// ---------------------------------------------------------------------------
extern "C" void kernel_launch(void* const* d_in, const int* in_sizes, int n_in,
                              void* d_out, int out_size, void* d_ws, size_t ws_size,
                              hipStream_t stream) {
    const float* features = (const float*)d_in[0];
    const int*   captions = (const int*)d_in[1];
    const int*   caplen   = (const int*)d_in[2];
    const float* emb  = (const float*)d_in[3];
    const float* Wea  = (const float*)d_in[4];
    const float* bea  = (const float*)d_in[5];
    const float* Wda  = (const float*)d_in[6];
    const float* bda  = (const float*)d_in[7];
    const float* Wfa  = (const float*)d_in[8];
    const float* bfa  = (const float*)d_in[9];
    const float* Wih  = (const float*)d_in[10];
    const float* bih  = (const float*)d_in[11];
    const float* Whh  = (const float*)d_in[12];
    const float* bhh  = (const float*)d_in[13];
    const float* Wh0  = (const float*)d_in[14];
    const float* bh0  = (const float*)d_in[15];
    const float* Wc0  = (const float*)d_in[16];
    const float* bc0  = (const float*)d_in[17];
    const float* Wfb  = (const float*)d_in[18];
    const float* bfb  = (const float*)d_in[19];
    const float* Wout = (const float*)d_in[20];
    const float* bout = (const float*)d_in[21];

    char* ws = (char*)d_ws;
    size_t off = 0;
    auto alloc = [&](size_t bytes) -> void* {
        off = (off + 255) & ~(size_t)255;
        void* p = ws + off;
        off += bytes;
        return p;
    };

    int*   sind   = (int*)alloc(BB * 4);
    int*   slen   = (int*)alloc(BB * 4);
    int*   caps_s = (int*)alloc(BB * TT * 4);
    float* enc    = (float*)alloc((size_t)BB * PP * DEE * 4);   // 51.4 MB
    float* meanE  = (float*)alloc((size_t)BB * DEE * 4);
    float* att1   = (float*)alloc((size_t)BB * PP * AAA * 4);   // 12.9 MB

    __hip_bfloat16* WeaT  = (__hip_bfloat16*)alloc((size_t)AAA * DEE * 2);
    __hip_bfloat16* WdaT  = (__hip_bfloat16*)alloc((size_t)AAA * HHH * 2);
    __hip_bfloat16* WfbT  = (__hip_bfloat16*)alloc((size_t)DEE * HHH * 2);
    __hip_bfloat16* Wh0T  = (__hip_bfloat16*)alloc((size_t)HHH * DEE * 2);
    __hip_bfloat16* Wc0T  = (__hip_bfloat16*)alloc((size_t)HHH * DEE * 2);
    __hip_bfloat16* WoutT = (__hip_bfloat16*)alloc((size_t)VVV * HHH * 2);  // 30.7 MB
    __hip_bfloat16* WihB  = (__hip_bfloat16*)alloc((size_t)G4 * XXX * 2);   // 10.5 MB
    __hip_bfloat16* WhhB  = (__hip_bfloat16*)alloc((size_t)G4 * HHH * 2);

    // packed A-fragment buffers (bf16, WMMA layout)
    __hip_bfloat16* packEnc  = (__hip_bfloat16*)alloc((size_t)BB * PP * DEE * 2); // 25.7 MB
    __hip_bfloat16* packMean = (__hip_bfloat16*)alloc((size_t)BB * DEE * 2);
    __hip_bfloat16* packH    = (__hip_bfloat16*)alloc((size_t)BB * HHH * 2);
    __hip_bfloat16* packX    = (__hip_bfloat16*)alloc((size_t)BB * XXX * 2);
    __hip_bfloat16* packHn   = (__hip_bfloat16*)alloc((size_t)BB * HHH * 2);

    float* hbuf  = (float*)alloc((size_t)BB * HHH * 4);
    float* cbuf  = (float*)alloc((size_t)BB * HHH * 4);
    float* hnew  = (float*)alloc((size_t)BB * HHH * 4);
    float* att2  = (float*)alloc((size_t)BB * AAA * 4);
    float* esc   = (float*)alloc((size_t)BB * PP * 4);
    float* awe   = (float*)alloc((size_t)BB * DEE * 4);
    float* gateb = (float*)alloc((size_t)BB * DEE * 4);
    float* xbuf  = (float*)alloc((size_t)BB * XXX * 4);
    float* gates = (float*)alloc((size_t)BB * G4 * 4);

    // output layout: predictions | alphas | caps | lengths | sort_ind (all f32)
    float* out      = (float*)d_out;
    float* outP     = out;
    float* outA     = outP + (size_t)BB * TDEC * VVV;
    float* outCaps  = outA + (size_t)BB * TDEC * PP;
    float* outLen   = outCaps + BB * TT;
    float* outSind  = outLen + BB;

    // ----- one-time prep -----
    k_sort<<<1, 1, 0, stream>>>(caplen, captions, sind, slen, caps_s,
                                outCaps, outLen, outSind);
    {
        size_t n = (size_t)BB * PP * DEE;
        k_gather_enc<<<(unsigned)((n + 255) / 256), 256, 0, stream>>>(features, sind, enc);
    }
    k_mean<<<BB, 256, 0, stream>>>(enc, meanE);

    auto cvt = [&](const float* s, __hip_bfloat16* d, size_t n) {
        k_cvt<<<(unsigned)((n + 255) / 256), 256, 0, stream>>>(s, d, (int)n);
    };
    auto tcvt = [&](const float* s, __hip_bfloat16* d, int R, int C) {
        size_t n = (size_t)R * C;
        k_tcvt<<<(unsigned)((n + 255) / 256), 256, 0, stream>>>(s, d, R, C);
    };
    auto packA = [&](const float* s, int lda, __hip_bfloat16* d, int M, int K) {
        size_t n = (size_t)M * K;
        k_packA<<<(unsigned)((n + 255) / 256), 256, 0, stream>>>(s, lda, d, M, K);
    };
    tcvt(Wea, WeaT, DEE, AAA);
    tcvt(Wda, WdaT, HHH, AAA);
    tcvt(Wfb, WfbT, HHH, DEE);
    tcvt(Wh0, Wh0T, DEE, HHH);
    tcvt(Wc0, Wc0T, DEE, HHH);
    tcvt(Wout, WoutT, HHH, VVV);
    cvt(Wih, WihB, (size_t)G4 * XXX);
    cvt(Whh, WhhB, (size_t)G4 * HHH);

    auto gemm = [&](const __hip_bfloat16* Ap, const __hip_bfloat16* Bp, const float* bi,
                    float* Cp, long long ldc, int M, int N, int K, int act, int accum,
                    const int* ml, int t) {
        int tiles = (M >> 4) * ((N + 63) >> 6);
        k_gemm<<<(tiles + 7) / 8, 256, 0, stream>>>(Ap, Bp, bi, Cp, ldc,
                                                    M, N, K, act, accum, ml, t);
    };

    packA(meanE, DEE, packMean, BB, DEE);
    packA(enc,   DEE, packEnc,  BB * PP, DEE);

    gemm(packMean, Wh0T, bh0, hbuf, HHH, BB, HHH, DEE, 0, 0, nullptr, 0);
    gemm(packMean, Wc0T, bc0, cbuf, HHH, BB, HHH, DEE, 0, 0, nullptr, 0);
    gemm(packEnc,  WeaT, bea, att1, AAA, BB * PP, AAA, DEE, 0, 0, nullptr, 0);

    // ----- decode loop -----
    for (int t = 0; t < TDEC; ++t) {
        packA(hbuf, HHH, packH, BB, HHH);
        gemm(packH, WdaT, bda, att2, AAA, BB, AAA, HHH, 0, 0, nullptr, 0);
        k_escore<<<(BB * PP + 7) / 8, 256, 0, stream>>>(att1, att2, Wfa, bfa, esc);
        k_softmax_awe<<<BB, 256, 0, stream>>>(esc, enc, awe, outA, slen, t);
        gemm(packH, WfbT, bfb, gateb, DEE, BB, DEE, HHH, 2, 0, nullptr, 0);
        k_build_x<<<(BB * XXX + 255) / 256, 256, 0, stream>>>(caps_s, emb, gateb, awe, xbuf, t);
        packA(xbuf, XXX, packX, BB, XXX);
        gemm(packX, WihB, bih, gates, G4, BB, G4, XXX, 0, 0, nullptr, 0);
        gemm(packH, WhhB, bhh, gates, G4, BB, G4, HHH, 0, 1, nullptr, 0);
        k_lstm<<<(BB * HHH + 255) / 256, 256, 0, stream>>>(gates, hbuf, cbuf, hnew, slen, t);
        packA(hnew, HHH, packHn, BB, HHH);
        gemm(packHn, WoutT, bout, outP + (size_t)t * VVV, (long long)TDEC * VVV,
             BB, VVV, HHH, 0, 0, slen, t);
    }
}
